// LinearEncoderBlock_43808666419725
// MI455X (gfx1250) — matrix operand order repaired and checked by reading, not compiled, not run on hardware
//
#include <hip/hip_runtime.h>
#include <stdint.h>

// ---------------------------------------------------------------------------
// LinearEncoderBlock for MI455X (gfx1250, wave32, WMMA bf16 16x16x32).
// v3: 128x256 block tile, 8 waves, 64x64 wave tile (16 accumulators),
// BK=64, double-buffered LDS fed by GLOBAL_LOAD_ASYNC_TO_LDS_B128 when the
// builtin is available (sync fallback otherwise).
// ---------------------------------------------------------------------------

typedef __attribute__((ext_vector_type(16))) __bf16 v16bf;
typedef __attribute__((ext_vector_type(8)))  __bf16 bf16x8;
typedef __attribute__((ext_vector_type(8)))  float  v8f;

#define BB 4
#define SS 2048
#define DD 2048
#define FF 8192

#define BM 128
#define BN 256
#define BK 64
#define NTHREADS 256

#if defined(__has_builtin)
#if __has_builtin(__builtin_amdgcn_global_load_async_to_lds_b128)
#define HAVE_ASYNC_LDS 1
#endif
#endif
#ifndef HAVE_ASYNC_LDS
#define HAVE_ASYNC_LDS 0
#endif

__device__ __forceinline__ uint16_t f2bf(float f) {
    union { float f; uint32_t u; } c; c.f = f;
    uint32_t u = c.u;
    return (uint16_t)((u + 0x7FFFu + ((u >> 16) & 1u)) >> 16);  // RNE
}
__device__ __forceinline__ float bf2f(uint16_t b) {
    union { uint32_t u; float f; } c; c.u = ((uint32_t)b) << 16;
    return c.f;
}
__device__ __forceinline__ float featmap(float x) {        // elu(x)+1
    return x > 0.f ? x + 1.f : __expf(x);
}

// 16-byte global -> LDS copy; async (ASYNCcnt-tracked) when supported.
// Builtin parameter types (from hipcc diagnostic): pointer to
// `int __attribute__((vector_size(16)))` in addrspace(1) / addrspace(3).
typedef int async_v4i __attribute__((vector_size(16)));

__device__ __forceinline__ void copy16_g2l(const uint16_t* g, uint16_t* l) {
#if HAVE_ASYNC_LDS
    __builtin_amdgcn_global_load_async_to_lds_b128(
        (__attribute__((address_space(1))) async_v4i*)(void*)const_cast<uint16_t*>(g),
        (__attribute__((address_space(3))) async_v4i*)(void*)l,
        0, 0);
#else
    *(uint4*)l = *(const uint4*)g;
#endif
}

__device__ __forceinline__ void wait_async_copies() {
#if HAVE_ASYNC_LDS
#if __has_builtin(__builtin_amdgcn_s_wait_asynccnt)
    __builtin_amdgcn_s_wait_asynccnt(0);
#else
    asm volatile("s_wait_asynccnt 0x0" ::: "memory");
#endif
#endif
}

// ---------------------------------------------------------------------------
// Tiled WMMA GEMM: C[M,N] = A[M,K] * B, fp32 accum, bf16 operands.
// TRANSB=true : B stored [N,K] (torch Linear weight / V-as-B).
// TRANSB=false: B stored [K,N] (kv matrix).
// 256 threads = 8 waves arranged 2x4; wave tile 64x64 = 4x4 WMMA subtiles.
// EPI: 0=store bf16   1=QKV split + feature map   2=row-scale -> bf16
//      3=bias+residual -> f32   4=relu(bias) -> bf16
// ---------------------------------------------------------------------------
template<int EPI, bool TRANSB>
__global__ __launch_bounds__(NTHREADS, 1)
void gemm_wmma(const uint16_t* __restrict__ Ag, const uint16_t* __restrict__ Bg,
               int M, int N, int K, int lda, int ldb,
               long batchA, long batchB,
               void* __restrict__ out0, void* __restrict__ out1,
               void* __restrict__ out2,
               long batchOut, int ldo,
               const float* __restrict__ bias,
               const float* __restrict__ resid, long batchRes,
               const float* __restrict__ rowscale, long batchScale)
{
    __shared__ __align__(16) uint16_t sA[2][BM * BK];   // [m][k]
    __shared__ __align__(16) uint16_t sB[2][BN * BK];   // [n][k]

    const int tid   = threadIdx.x;
    const int lane  = tid & 31;
    const int wid   = tid >> 5;
    const int widm  = wid >> 2;           // 0..1
    const int widn  = wid & 3;            // 0..3
    const int zb    = blockIdx.z;
    const int m0    = blockIdx.y * BM;
    const int n0    = blockIdx.x * BN;
    const int mlane = lane & 15;
    const int ahalf = lane >> 4;          // 0: lanes 0-15, 1: lanes 16-31

    Ag += (long)zb * batchA;
    Bg += (long)zb * batchB;

    v8f acc[4][4];
    {
        v8f vz = {};
        #pragma unroll
        for (int i = 0; i < 4; i++)
            #pragma unroll
            for (int j = 0; j < 4; j++) acc[i][j] = vz;
    }

    union Frag { v16bf v; bf16x8 h[2]; };

    // ---- tile staging (per-thread slices) ----
    auto stage_A = [&](uint16_t* buf, int kb) {
        int row = tid >> 1;                 // 128 rows / 2 threads each
        int ko  = (tid & 1) * 32;           // 32 elems = 64 B per thread
        const uint16_t* g = Ag + (long)(m0 + row) * lda + kb + ko;
        uint16_t* l = buf + row * BK + ko;
        #pragma unroll
        for (int i = 0; i < 4; i++) copy16_g2l(g + i * 8, l + i * 8);
    };
    auto stage_B = [&](uint16_t* buf, int kb) {
        if (TRANSB) {
            const uint16_t* g = Bg + (long)(n0 + tid) * ldb + kb;   // 256 rows
            uint16_t* l = buf + tid * BK;                           // 64 elems
            #pragma unroll
            for (int i = 0; i < 8; i++) copy16_g2l(g + i * 8, l + i * 8);
        } else {
            int k  = tid >> 2;              // 64 k-rows / 4 threads each
            int nb = (tid & 3) * 64;        // 64 n-elems per thread
            const uint16_t* g = Bg + (long)(kb + k) * ldb + n0 + nb;
            for (int j = 0; j < 64; j++) buf[(nb + j) * BK + k] = g[j];
        }
    };

    // ---- prologue: fill buffer 0 ----
    stage_A(sA[0], 0);
    stage_B(sB[0], 0);
    wait_async_copies();
    __syncthreads();

    int cur = 0;
    for (int kb = 0; kb < K; kb += BK) {
        const int nxt = cur ^ 1;
        if (kb + BK < K) {
            stage_A(sA[nxt], kb + BK);
            stage_B(sB[nxt], kb + BK);
            // hint the tile after next toward the caches
            __builtin_prefetch(Ag + (long)(m0 + (tid >> 1)) * lda + kb + 2 * BK, 0, 1);
        }

        // ---- compute from current buffer: 2 k-steps of 16 WMMAs ----
        const uint16_t* a = sA[cur];
        const uint16_t* b = sB[cur];
        #pragma unroll
        for (int kk = 0; kk < BK; kk += 32) {
            // A 16x32 frag: lanes<16 hold K 0-7/16-23, lanes>=16 K 8-15/24-31.
            Frag af[4];
            #pragma unroll
            for (int mi = 0; mi < 4; mi++) {
                int row = widm * 64 + mi * 16 + mlane;
                int k1  = kk + ahalf * 8;
                af[mi].h[0] = *(const bf16x8*)(a + row * BK + k1);
                af[mi].h[1] = *(const bf16x8*)(a + row * BK + k1 + 16);
            }
            // B 32x16 frag: col n = lane&15; lanes<16 K 0-7/16-23, lanes>=16 K 8-15/24-31.
            Frag bfr[4];
            #pragma unroll
            for (int ni = 0; ni < 4; ni++) {
                int col = widn * 64 + ni * 16 + mlane;
                int k1  = kk + ahalf * 8;
                bfr[ni].h[0] = *(const bf16x8*)(b + col * BK + k1);
                bfr[ni].h[1] = *(const bf16x8*)(b + col * BK + k1 + 16);
            }
            #pragma unroll
            for (int mi = 0; mi < 4; mi++)
                #pragma unroll
                for (int ni = 0; ni < 4; ni++)
                    acc[mi][ni] = __builtin_amdgcn_wmma_f32_16x16x32_bf16(
                        false, af[mi].v, false, bfr[ni].v,
                        (short)0, acc[mi][ni], false, false);
        }

        wait_async_copies();
        __syncthreads();
        cur = nxt;
    }

    // ---- epilogue: D layout: m = r + 8*(lane>>4), n = lane&15 per 16x16 ----
    #pragma unroll
    for (int mi = 0; mi < 4; mi++) {
        #pragma unroll
        for (int ni = 0; ni < 4; ni++) {
            #pragma unroll
            for (int r = 0; r < 8; r++) {
                int m = m0 + widm * 64 + mi * 16 + ahalf * 8 + r;
                int n = n0 + widn * 64 + ni * 16 + mlane;
                float v = acc[mi][ni][r];
                if (EPI == 0) {
                    ((uint16_t*)out0)[(long)zb * batchOut + (long)m * ldo + n] = f2bf(v);
                } else if (EPI == 1) {
                    float t = featmap(v);
                    if (n < DD)
                        ((uint16_t*)out0)[(long)m * DD + n] = f2bf(t);            // q
                    else if (n < 2 * DD)
                        ((uint16_t*)out1)[(long)m * DD + (n - DD)] = f2bf(t);     // k
                    else
                        ((uint16_t*)out2)[(long)m * DD + (n - 2 * DD)] = f2bf(v); // v
                } else if (EPI == 2) {
                    float sc = rowscale[(long)zb * batchScale + m];
                    ((uint16_t*)out0)[(long)zb * batchOut + (long)m * ldo + n] = f2bf(v * sc);
                } else if (EPI == 3) {
                    float o = v + bias[n] + resid[(long)zb * batchRes + (long)m * ldo + n];
                    ((float*)out0)[(long)zb * batchOut + (long)m * ldo + n] = o;
                } else if (EPI == 4) {
                    float o = v + bias[n];
                    o = o > 0.f ? o : 0.f;
                    ((uint16_t*)out0)[(long)zb * batchOut + (long)m * ldo + n] = f2bf(o);
                }
            }
        }
    }
}

// ---------------------------------------------------------------------------
// Small helper kernels (bandwidth trivial)
// ---------------------------------------------------------------------------
__global__ void f32_to_bf16_kernel(const float* __restrict__ in,
                                   uint16_t* __restrict__ out, long n) {
    long i = (long)blockIdx.x * blockDim.x + threadIdx.x;
    long stride = (long)gridDim.x * blockDim.x;
    for (; i < n; i += stride) out[i] = f2bf(in[i]);
}

// k_sum[b,d] = sum_s K[b,s,d]
__global__ void ksum_kernel(const uint16_t* __restrict__ Kb, float* __restrict__ ks) {
    int d = blockIdx.x * blockDim.x + threadIdx.x;
    int b = blockIdx.y;
    const uint16_t* p = Kb + (long)b * SS * DD + d;
    float acc = 0.f;
    for (int s = 0; s < SS; s++) acc += bf2f(p[(long)s * DD]);
    ks[b * DD + d] = acc;
}

// z[b,i] = 1 / (sum_j Q[b,j,i] * ksum[b,j] + eps)
__global__ void zvec_kernel(const uint16_t* __restrict__ Qb,
                            const float* __restrict__ ks, float* __restrict__ z) {
    int i = blockIdx.x * blockDim.x + threadIdx.x;
    int b = blockIdx.y;
    const uint16_t* p  = Qb + (long)b * SS * DD + i;
    const float*    kv = ks + b * DD;
    float acc = 0.f;
    for (int j = 0; j < SS; j++) acc += bf2f(p[(long)j * DD]) * kv[j];
    z[b * DD + i] = 1.f / (acc + 1e-6f);
}

// LayerNorm over last dim (D=2048); one block per row. Optionally emits bf16.
__global__ __launch_bounds__(256)
void layernorm_kernel(const float* __restrict__ in, const float* __restrict__ g,
                      const float* __restrict__ be, float* __restrict__ outf,
                      uint16_t* __restrict__ outb) {
    __shared__ float red[256];
    long row = blockIdx.x;
    const float* x = in + row * DD;
    int tid = threadIdx.x;

    float s = 0.f;
    for (int i = tid; i < DD; i += 256) s += x[i];
    red[tid] = s; __syncthreads();
    for (int o = 128; o > 0; o >>= 1) { if (tid < o) red[tid] += red[tid + o]; __syncthreads(); }
    float mu = red[0] * (1.f / DD);
    __syncthreads();

    float v = 0.f;
    for (int i = tid; i < DD; i += 256) { float d = x[i] - mu; v += d * d; }
    red[tid] = v; __syncthreads();
    for (int o = 128; o > 0; o >>= 1) { if (tid < o) red[tid] += red[tid + o]; __syncthreads(); }
    float rstd = rsqrtf(red[0] * (1.f / DD) + 1e-5f);

    for (int i = tid; i < DD; i += 256) {
        float y = (x[i] - mu) * rstd * g[i] + be[i];
        outf[row * DD + i] = y;
        if (outb) outb[row * DD + i] = f2bf(y);
    }
}

// ---------------------------------------------------------------------------
extern "C" void kernel_launch(void* const* d_in, const int* in_sizes, int n_in,
                              void* d_out, int out_size, void* d_ws, size_t ws_size,
                              hipStream_t stream) {
    (void)in_sizes; (void)n_in; (void)out_size; (void)ws_size;

    const float* x     = (const float*)d_in[0];
    const float* w_qkv = (const float*)d_in[1];
    const float* w_o   = (const float*)d_in[2];
    const float* b_o   = (const float*)d_in[3];
    const float* w1    = (const float*)d_in[4];
    const float* b1    = (const float*)d_in[5];
    const float* w2    = (const float*)d_in[6];
    const float* b2    = (const float*)d_in[7];
    const float* g1    = (const float*)d_in[8];
    const float* be1   = (const float*)d_in[9];
    const float* g2    = (const float*)d_in[10];
    const float* be2   = (const float*)d_in[11];

    char* wp = (char*)d_ws;
    auto alloc = [&](size_t bytes) {
        char* p = wp; wp += (bytes + 255) & ~(size_t)255; return (void*)p;
    };
    const size_t BSD = (size_t)BB * SS * DD;
    const size_t BSS = (size_t)BB * SS * SS;
    const size_t BSF = (size_t)BB * SS * FF;

    uint16_t* xb    = (uint16_t*)alloc(BSD * 2);
    uint16_t* wqkvb = (uint16_t*)alloc((size_t)3 * DD * DD * 2);
    uint16_t* wob   = (uint16_t*)alloc((size_t)DD * DD * 2);
    uint16_t* w1b   = (uint16_t*)alloc((size_t)FF * DD * 2);
    uint16_t* w2b   = (uint16_t*)alloc((size_t)FF * DD * 2);
    uint16_t* Qb    = (uint16_t*)alloc(BSD * 2);
    uint16_t* Kb    = (uint16_t*)alloc(BSD * 2);
    uint16_t* Vb    = (uint16_t*)alloc(BSD * 2);
    uint16_t* kvb   = (uint16_t*)alloc(BSS * 2);
    uint16_t* attnb = (uint16_t*)alloc(BSS * 2);
    float*    ksum  = (float*)alloc((size_t)BB * DD * 4);
    float*    zvec  = (float*)alloc((size_t)BB * DD * 4);
    float*    res1  = (float*)alloc(BSD * 4);
    float*    x2f   = (float*)alloc(BSD * 4);
    uint16_t* x2b   = (uint16_t*)alloc(BSD * 2);
    uint16_t* hbuf  = (uint16_t*)alloc(BSF * 2);
    float*    res2  = (float*)alloc(BSD * 4);

    // fp32 -> bf16 conversions
    f32_to_bf16_kernel<<<2048, 256, 0, stream>>>(x,     xb,    (long)BSD);
    f32_to_bf16_kernel<<<2048, 256, 0, stream>>>(w_qkv, wqkvb, (long)3 * DD * DD);
    f32_to_bf16_kernel<<<2048, 256, 0, stream>>>(w_o,   wob,   (long)DD * DD);
    f32_to_bf16_kernel<<<2048, 256, 0, stream>>>(w1,    w1b,   (long)FF * DD);
    f32_to_bf16_kernel<<<2048, 256, 0, stream>>>(w2,    w2b,   (long)FF * DD);

    // GEMM1: qkv = x @ w_qkv^T, split + feature map   [8192 x 6144, K=2048]
    gemm_wmma<1, true><<<dim3(3 * DD / BN, BB * SS / BM, 1), NTHREADS, 0, stream>>>(
        xb, wqkvb, BB * SS, 3 * DD, DD, DD, DD, 0, 0,
        Qb, Kb, Vb, 0, DD, nullptr, nullptr, 0, nullptr, 0);

    // k_sum and z
    ksum_kernel<<<dim3(DD / 256, BB), 256, 0, stream>>>(Kb, ksum);
    zvec_kernel<<<dim3(DD / 256, BB), 256, 0, stream>>>(Qb, ksum, zvec);

    // GEMM2: kv[b] = K[b] @ V[b]^T   (batched, B as [N,K])
    gemm_wmma<0, true><<<dim3(SS / BN, SS / BM, BB), NTHREADS, 0, stream>>>(
        Kb, Vb, SS, SS, DD, DD, DD, (long)SS * DD, (long)SS * DD,
        kvb, nullptr, nullptr, (long)SS * SS, SS, nullptr, nullptr, 0, nullptr, 0);

    // GEMM3: attn[b] = (Q[b] @ kv[b]) * z[b,row]   (B as [K,N])
    gemm_wmma<2, false><<<dim3(SS / BN, SS / BM, BB), NTHREADS, 0, stream>>>(
        Qb, kvb, SS, SS, DD, DD, SS, (long)SS * DD, (long)SS * SS,
        attnb, nullptr, nullptr, (long)SS * SS, SS, nullptr, nullptr, 0, zvec, DD);

    // GEMM4: res1[b] = attn[b] @ w_o^T + b_o + x[b]
    gemm_wmma<3, true><<<dim3(DD / BN, SS / BM, BB), NTHREADS, 0, stream>>>(
        attnb, wob, SS, DD, SS, SS, DD, (long)SS * SS, 0,
        res1, nullptr, nullptr, (long)SS * DD, DD, b_o, x, (long)SS * DD, nullptr, 0);

    // LN1 -> x2 (fp32 for residual, bf16 for FFN GEMM)
    layernorm_kernel<<<BB * SS, 256, 0, stream>>>(res1, g1, be1, x2f, x2b);

    // GEMM5: h = relu(x2 @ w1^T + b1)   [8192 x 8192, K=2048]
    gemm_wmma<4, true><<<dim3(FF / BN, BB * SS / BM, 1), NTHREADS, 0, stream>>>(
        x2b, w1b, BB * SS, FF, DD, DD, DD, 0, 0,
        hbuf, nullptr, nullptr, 0, FF, b1, nullptr, 0, nullptr, 0);

    // GEMM6: res2 = h @ w2^T + b2 + x2   [8192 x 2048, K=8192]
    gemm_wmma<3, true><<<dim3(DD / BN, BB * SS / BM, 1), NTHREADS, 0, stream>>>(
        hbuf, w2b, BB * SS, DD, FF, FF, FF, 0, 0,
        res2, nullptr, nullptr, 0, DD, b2, x2f, 0, nullptr, 0);

    // LN2 -> output (fp32)
    layernorm_kernel<<<BB * SS, 256, 0, stream>>>(res2, g2, be2, (float*)d_out, nullptr);
}
